// DeformCNN_25924422599130
// MI455X (gfx1250) — matrix operand order repaired and checked by reading, not compiled, not run on hardware
//
#include <hip/hip_runtime.h>
#include <hip/hip_bf16.h>

typedef __attribute__((ext_vector_type(16))) _Float16 v16h;
typedef __attribute__((ext_vector_type(8)))  _Float16 v8h;
typedef __attribute__((ext_vector_type(8)))  float    v8f;

__device__ __forceinline__ int iclamp(int v, int lo, int hi) {
    return v < lo ? lo : (v > hi ? hi : v);
}

// ---------------------------------------------------------------------------
// pack fp32 [Mr x Kr] row-major  ->  f16 [Mp x Kp] row-major, zero padded
// ---------------------------------------------------------------------------
__global__ void pack_f16(const float* __restrict__ src, _Float16* __restrict__ dst,
                         int Mr, int Kr, int Mp, int Kp) {
    int idx = blockIdx.x * blockDim.x + threadIdx.x;
    int total = Mp * Kp;
    if (idx >= total) return;
    int m = idx / Kp, k = idx % Kp;
    _Float16 v = (_Float16)0.f;
    if (m < Mr && k < Kr) v = (_Float16)src[(size_t)m * Kr + k];
    dst[idx] = v;
}

// ---------------------------------------------------------------------------
// im2col for 5x5 pad-2 conv: src (256,C,H,W) fp32 -> Bcol f16 [N x Kp],
// column n = b*H*W + y*W + x contiguous over k; k = c*25 + ky*5 + kx, padded.
// ---------------------------------------------------------------------------
__global__ void im2col5x5(const float* __restrict__ src, _Float16* __restrict__ Bc,
                          int C, int H, int W, int Kp) {
    long tid = (long)blockIdx.x * blockDim.x + threadIdx.x;
    int  S = H * W;
    long total = (long)256 * S * Kp;
    if (tid >= total) return;
    int  k = (int)(tid % Kp);
    long n = tid / Kp;
    int  b = (int)(n / S);
    int  s = (int)(n % S);
    int  y = s / W, x = s % W;
    _Float16 v = (_Float16)0.f;
    int KR = C * 25;
    if (k < KR) {
        int c = k / 25, kk = k % 25;
        int ky = kk / 5, kx = kk % 5;
        int iy = y + ky - 2, ix = x + kx - 2;
        if (iy >= 0 && iy < H && ix >= 0 && ix < W)
            v = (_Float16)src[((size_t)(b * C + c) * H + iy) * W + ix];
    }
    Bc[(size_t)n * Kp + k] = v;
}

// ---------------------------------------------------------------------------
// Deformable sampling stage 1: x (256,1,28,28), off1 (256,50,28,28)
//   -> Bcol f16 [200704 x 32], row k = kernel tap kk (C=1), rows 25..31 = 0
// ---------------------------------------------------------------------------
__global__ void dsample1(const float* __restrict__ x, const float* __restrict__ off,
                         _Float16* __restrict__ Bc) {
    int tid = blockIdx.x * blockDim.x + threadIdx.x;
    if (tid >= 200704 * 32) return;
    int k = tid & 31;
    int n = tid >> 5;
    _Float16 v = (_Float16)0.f;
    if (k < 25) {
        int b = n / 784, s = n % 784;
        int y = s / 28, xx = s % 28;
        const float* ob = off + (size_t)b * 50 * 784;
        float dy = ob[(size_t)(k * 2)     * 784 + s];
        float dx = ob[(size_t)(k * 2 + 1) * 784 + s];
        int ky = k / 5, kx = k % 5;
        float py = dy + (float)(y  + ky - 2);
        float px = dx + (float)(xx + kx - 2);
        float y0f = floorf(py), x0f = floorf(px);
        float ly = py - y0f, lx = px - x0f;
        int y0 = (int)y0f, x0 = (int)x0f;
        const float* xb = x + (size_t)b * 784;
        float acc = 0.f;
#pragma unroll
        for (int dyi = 0; dyi < 2; ++dyi)
#pragma unroll
            for (int dxi = 0; dxi < 2; ++dxi) {
                int yi = y0 + dyi, xi = x0 + dxi;
                float wgt = (dyi ? ly : 1.f - ly) * (dxi ? lx : 1.f - lx);
                float valid = (yi >= 0 && yi < 28 && xi >= 0 && xi < 28) ? 1.f : 0.f;
                acc += xb[iclamp(yi, 0, 27) * 28 + iclamp(xi, 0, 27)] * valid * wgt;
            }
        v = (_Float16)acc;
    }
    Bc[(size_t)n * 32 + k] = v;
}

// ---------------------------------------------------------------------------
// Deformable sampling stage 2: p1 (256,16,14,14), off2 (256,50,14,14)
//   -> Bcol f16 [50176 x 416], row k = c*25 + kk; rows 400..415 = 0
// one thread handles one (column, tap) and all 16 channels; j==25 writes pad
// ---------------------------------------------------------------------------
__global__ void dsample2(const float* __restrict__ p1, const float* __restrict__ off,
                         _Float16* __restrict__ Bc) {
    int tid = blockIdx.x * blockDim.x + threadIdx.x;
    if (tid >= 50176 * 26) return;
    int j = tid % 26;
    int n = tid / 26;
    if (j == 25) {
        _Float16* d = Bc + (size_t)n * 416 + 400;
#pragma unroll
        for (int t = 0; t < 16; ++t) d[t] = (_Float16)0.f;
        return;
    }
    int b = n / 196, s = n % 196;
    int y = s / 14, xx = s % 14;
    const float* ob = off + (size_t)b * 50 * 196;
    float dy = ob[(size_t)(j * 2)     * 196 + s];
    float dx = ob[(size_t)(j * 2 + 1) * 196 + s];
    int ky = j / 5, kx = j % 5;
    float py = dy + (float)(y  + ky - 2);
    float px = dx + (float)(xx + kx - 2);
    float y0f = floorf(py), x0f = floorf(px);
    float ly = py - y0f, lx = px - x0f;
    int y0 = (int)y0f, x0 = (int)x0f;
    int yc0 = iclamp(y0, 0, 13), yc1 = iclamp(y0 + 1, 0, 13);
    int xc0 = iclamp(x0, 0, 13), xc1 = iclamp(x0 + 1, 0, 13);
    float v00 = (y0 >= 0     && y0 < 14     && x0 >= 0     && x0 < 14    ) ? 1.f : 0.f;
    float v01 = (y0 >= 0     && y0 < 14     && x0 + 1 >= 0 && x0 + 1 < 14) ? 1.f : 0.f;
    float v10 = (y0 + 1 >= 0 && y0 + 1 < 14 && x0 >= 0     && x0 < 14    ) ? 1.f : 0.f;
    float v11 = (y0 + 1 >= 0 && y0 + 1 < 14 && x0 + 1 >= 0 && x0 + 1 < 14) ? 1.f : 0.f;
    float w00 = (1.f - ly) * (1.f - lx) * v00;
    float w01 = (1.f - ly) * lx         * v01;
    float w10 = ly         * (1.f - lx) * v10;
    float w11 = ly         * lx         * v11;
    _Float16* dst = Bc + (size_t)n * 416;
    const float* pb = p1 + (size_t)b * 16 * 196;
#pragma unroll
    for (int c = 0; c < 16; ++c) {
        const float* pc = pb + c * 196;
        float sv = pc[yc0 * 14 + xc0] * w00 + pc[yc0 * 14 + xc1] * w01
                 + pc[yc1 * 14 + xc0] * w10 + pc[yc1 * 14 + xc1] * w11;
        dst[c * 25 + j] = (_Float16)sv;
    }
}

// ---------------------------------------------------------------------------
// relu + 2x2 maxpool: in (BC,H,W) -> out (BC,H/2,W/2)
// ---------------------------------------------------------------------------
__global__ void relu_maxpool2(const float* __restrict__ in, float* __restrict__ out,
                              int BC, int H, int W) {
    int Ho = H >> 1, Wo = W >> 1;
    int idx = blockIdx.x * blockDim.x + threadIdx.x;
    int total = BC * Ho * Wo;
    if (idx >= total) return;
    int wo = idx % Wo;
    int t  = idx / Wo;
    int ho = t % Ho;
    int bc = t / Ho;
    const float* p = in + ((size_t)bc * H + ho * 2) * W + wo * 2;
    float m = fmaxf(fmaxf(p[0], p[1]), fmaxf(p[W], p[W + 1]));
    out[idx] = fmaxf(m, 0.f);
}

// ---------------------------------------------------------------------------
// Generic WMMA GEMM:  out[m,n] = sum_k A[m,k]*B[k,n] + bias[m]
//   A: f16 [Mp x Kp] row-major (Mp mult of 16, Kp mult of 32, zero padded)
//   B: f16 [N cols x Kp] column-major (column contiguous)
//   out fp32, written as out[((n/S)*Mreal + m)*S + (n%S)] for m < Mreal
// One wave computes one 16x16 tile with a chain of v_wmma_f32_16x16x32_f16.
// Per-lane fragments are contiguous 16-k slices (this code's order), so A = one v16h
// load and B = one B256-equivalent (v16h) load per lane, no LDS needed.
// ---------------------------------------------------------------------------
__global__ void gemm_wmma(const _Float16* __restrict__ A, const _Float16* __restrict__ B,
                          const float* __restrict__ bias, float* __restrict__ out,
                          int Mp, int Mreal, int N, int Kp, int S) {
    int wave = (blockIdx.x * blockDim.x + threadIdx.x) >> 5;
    int lane = threadIdx.x & 31;
    int mtiles = Mp >> 4;
    int ntiles = (N + 15) >> 4;
    if (wave >= mtiles * ntiles) return;          // wave-uniform exit: EXEC all-ones below
    int mt = wave % mtiles;
    int nt = wave / mtiles;
    int m0 = mt << 4, n0 = nt << 4;
    int g = lane >> 4;                            // lane group (0: lanes 0-15, 1: 16-31)
    int l = lane & 15;
    const _Float16* Arow = A + (size_t)(m0 + l) * Kp;
    int ncol = n0 + l;
    if (ncol >= N) ncol = N - 1;                  // clamp reads; writes guarded below
    const _Float16* Bcol = B + (size_t)ncol * Kp;
    v8f acc = {};
    for (int kc = 0; kc < Kp; kc += 32) {
        // A fragment: lane(g,m): K = kc+16g .. +15, contiguous (same k order as B)
        v16h a = *(const v16h*)(Arow + kc + 16 * g);
        // B fragment (32x16 layout): lane(g,n): K = kc+16g .. +15, contiguous
        v16h b = *(const v16h*)(Bcol + kc + 16 * g);
        acc = __builtin_amdgcn_wmma_f32_16x16x32_f16(false, a, false, b,
                                                     (short)0, acc, false, false);
    }
    int n = n0 + l;
    if (n < N) {
        int bq = n / S, sr = n % S;
#pragma unroll
        for (int r = 0; r < 8; ++r) {
            int m = m0 + r + 8 * g;               // C/D layout: VGPR r, group g -> M = r+8g
            if (m < Mreal)
                out[((size_t)bq * Mreal + m) * S + sr] = acc[r] + bias[m];
        }
    }
}

// ---------------------------------------------------------------------------
extern "C" void kernel_launch(void* const* d_in, const int* in_sizes, int n_in,
                              void* d_out, int out_size, void* d_ws, size_t ws_size,
                              hipStream_t stream) {
    (void)in_sizes; (void)n_in; (void)out_size; (void)ws_size;
    const float* x      = (const float*)d_in[0];
    const float* w_off1 = (const float*)d_in[1];
    const float* b_off1 = (const float*)d_in[2];
    const float* w_dc1  = (const float*)d_in[3];
    const float* b_dc1  = (const float*)d_in[4];
    const float* w_off2 = (const float*)d_in[5];
    const float* b_off2 = (const float*)d_in[6];
    const float* w_dc2  = (const float*)d_in[7];
    const float* b_dc2  = (const float*)d_in[8];
    const float* w_fc   = (const float*)d_in[9];
    const float* b_fc   = (const float*)d_in[10];
    float* out = (float*)d_out;

    // scratch sub-allocation (256B aligned)
    char* base = (char*)d_ws;
    size_t off = 0;
    auto alloc = [&](size_t bytes) -> void* {
        off = (off + 255) & ~(size_t)255;
        void* p = base + off;
        off += bytes;
        return p;
    };
    _Float16* A_off1 = (_Float16*)alloc((size_t)64 * 32 * 2);
    _Float16* A_dc1  = (_Float16*)alloc((size_t)16 * 32 * 2);
    _Float16* A_off2 = (_Float16*)alloc((size_t)64 * 416 * 2);
    _Float16* A_dc2  = (_Float16*)alloc((size_t)32 * 416 * 2);
    _Float16* A_fc   = (_Float16*)alloc((size_t)16 * 1568 * 2);
    _Float16* Bc1 = (_Float16*)alloc((size_t)200704 * 32 * 2);   // stage1 im2col / deform samples
    _Float16* Bc2 = (_Float16*)alloc((size_t)50176 * 416 * 2);   // stage2 im2col / deform samples
    _Float16* Bfc = (_Float16*)alloc((size_t)256 * 1568 * 2);
    float* buf1 = (float*)alloc((size_t)10035200 * 4);           // off1 (256,50,28,28) then h1
    float* p1   = (float*)alloc((size_t)802816  * 4);            // (256,16,14,14)
    float* buf2 = (float*)alloc((size_t)2508800 * 4);            // off2 (256,50,14,14) then h2
    float* p2   = (float*)alloc((size_t)401408  * 4);            // (256,32,7,7)

    auto cdiv = [](long a, long b) { return (int)((a + b - 1) / b); };
    const int T = 256;  // 8 waves per block

    // --- weight packing (fp32 -> padded f16) ---
    pack_f16<<<cdiv(64 * 32, T),   T, 0, stream>>>(w_off1, A_off1, 50, 25,   64, 32);
    pack_f16<<<cdiv(16 * 32, T),   T, 0, stream>>>(w_dc1,  A_dc1,  16, 25,   16, 32);
    pack_f16<<<cdiv(64 * 416, T),  T, 0, stream>>>(w_off2, A_off2, 50, 400,  64, 416);
    pack_f16<<<cdiv(32 * 416, T),  T, 0, stream>>>(w_dc2,  A_dc2,  32, 400,  32, 416);
    pack_f16<<<cdiv(16 * 1568, T), T, 0, stream>>>(w_fc,   A_fc,   10, 1568, 16, 1568);

    // --- stage 1: conv_off1 -> deform_conv1 -> pool ---
    im2col5x5<<<cdiv((long)200704 * 32, T), T, 0, stream>>>(x, Bc1, 1, 28, 28, 32);
    gemm_wmma<<<cdiv((long)(64 / 16) * (200704 / 16), 8), T, 0, stream>>>(
        A_off1, Bc1, b_off1, buf1, 64, 50, 200704, 32, 784);
    dsample1<<<cdiv((long)200704 * 32, T), T, 0, stream>>>(x, buf1, Bc1);
    gemm_wmma<<<cdiv((long)(16 / 16) * (200704 / 16), 8), T, 0, stream>>>(
        A_dc1, Bc1, b_dc1, buf1, 16, 16, 200704, 32, 784);
    relu_maxpool2<<<cdiv(256 * 16 * 14 * 14, T), T, 0, stream>>>(buf1, p1, 256 * 16, 28, 28);

    // --- stage 2: conv_off2 -> deform_conv2 -> pool ---
    im2col5x5<<<cdiv((long)50176 * 416, T), T, 0, stream>>>(p1, Bc2, 16, 14, 14, 416);
    gemm_wmma<<<cdiv((long)(64 / 16) * (50176 / 16), 8), T, 0, stream>>>(
        A_off2, Bc2, b_off2, buf2, 64, 50, 50176, 416, 196);
    dsample2<<<cdiv((long)50176 * 26, T), T, 0, stream>>>(p1, buf2, Bc2);
    gemm_wmma<<<cdiv((long)(32 / 16) * (50176 / 16), 8), T, 0, stream>>>(
        A_dc2, Bc2, b_dc2, buf2, 32, 32, 50176, 416, 196);
    relu_maxpool2<<<cdiv(256 * 32 * 7 * 7, T), T, 0, stream>>>(buf2, p2, 256 * 32, 14, 14);

    // --- fc: (256,1568) @ (1568,10)^T + b ---
    pack_f16<<<cdiv(256 * 1568, T), T, 0, stream>>>(p2, Bfc, 256, 1568, 256, 1568);
    gemm_wmma<<<cdiv((long)1 * (256 / 16), 8), T, 0, stream>>>(
        A_fc, Bfc, b_fc, out, 16, 10, 256, 1568, 1);
}
